// BackboneDenoisingLayer_49959059587659
// MI455X (gfx1250) — compile-verified, matrix operand order checked by reading, not run on hardware
//
#include <hip/hip_runtime.h>
#include <hip/hip_bf16.h>

// ---------------- constants ----------------
constexpr int NN = 16384;     // nodes
constexpr int EE = 262144;    // spatial edges
constexpr int ES = 131072;    // sequence edges
constexpr int GKS = 256;      // K staging tile (multiple of 32)

typedef __attribute__((ext_vector_type(16))) __bf16 v16bf;
typedef __attribute__((ext_vector_type(8)))  float  v8f;

__device__ __forceinline__ unsigned short f2bf(float f) {
  unsigned int u = __float_as_uint(f);
  unsigned int r = u + 0x7FFFu + ((u >> 16) & 1u);
  return (unsigned short)(r >> 16);
}
__device__ __forceinline__ float bf2f(unsigned short h) {
  return __uint_as_float(((unsigned int)h) << 16);
}
__device__ __forceinline__ unsigned int encf(float f) {
  unsigned int u = __float_as_uint(f);
  return (u & 0x80000000u) ? ~u : (u | 0x80000000u);
}
__device__ __forceinline__ float decf(unsigned int u) {
  u = (u & 0x80000000u) ? (u & 0x7FFFFFFFu) : ~u;
  return __uint_as_float(u);
}

// ---------------- generic WMMA GEMM ----------------
struct ASpec {
  const void* s0; const void* s1; const void* s2;
  const int* i0; const int* i1; const int* i2;
  int w0, w1, w2;      // segment widths (0 = unused)
  int t0, t1, t2;      // 0=fp32 source, 1=bf16 source
};
struct ESpec {
  const float* bias;
  const float* resid;      // [M,Nout], direct row
  const float* rscale;     // per-row scale applied before resid (mask)
  const float* ln_g; const float* ln_b;
  float* out_f;
  unsigned short* out_h;
  int relu;
};

__device__ __forceinline__ float seg_val(const void* s, int ty, const int* ix,
                                         int row, int w, int off) {
  int r = ix ? ix[row] : row;
  size_t p = (size_t)r * w + off;
  return ty ? bf2f(((const unsigned short*)s)[p]) : ((const float*)s)[p];
}
__device__ __forceinline__ float fetchA(const ASpec& A, int row, int k) {
  if (k < A.w0) return seg_val(A.s0, A.t0, A.i0, row, A.w0, k);
  k -= A.w0;
  if (k < A.w1) return seg_val(A.s1, A.t1, A.i1, row, A.w1, k);
  k -= A.w1;
  if (k < A.w2) return seg_val(A.s2, A.t2, A.i2, row, A.w2, k);
  return 0.f;
}

// Block = 128 threads (4 wave32), one 16-row output tile per block.
// A staged to LDS in WMMA per-lane packed bf16 layout; B pre-packed in global.
// NOUT templated so per-wave accumulators are compile-time indexed (VGPRs, no scratch).
template <int NOUT>
__global__ __launch_bounds__(128) void k_gemm(ASpec A, ESpec E_, const unsigned short* Wp,
                                              int M, int Kpad) {
  constexpr int NCH = NOUT / 16;         // 16-col chunks
  constexpr int NCPW = (NCH + 3) / 4;    // chunks per wave

  extern __shared__ char smem[];
  unsigned short* sA = (unsigned short*)smem;
  float* sC   = (float*)(smem + GKS * 16 * 2);
  float* sSum = sC + 16 * NOUT;
  float* sSq  = sSum + 16 * 8;
  float* sMean = sSq + 16 * 8;
  float* sInv  = sMean + 16;

  const int tid = threadIdx.x;
  const int wave = (tid >> 5) & 3, lane = tid & 31;
  const int row0 = blockIdx.x * 16;

  v8f acc[NCPW];
#pragma unroll
  for (int c = 0; c < NCPW; ++c) acc[c] = (v8f){0.f,0.f,0.f,0.f,0.f,0.f,0.f,0.f};

  for (int k0 = 0; k0 < Kpad; k0 += GKS) {
    int ks = Kpad - k0; if (ks > GKS) ks = GKS;
    for (int t = tid; t < 16 * ks; t += 128) {
      int m = t / ks, kk = t % ks;
      float v = fetchA(A, row0 + m, k0 + kk);
      int kl = kk & 31, kc = kk >> 5;
      int ln = m + (((kl >> 3) & 1) << 4);
      int idx = (kl & 7) + ((kl >> 4) << 3);
      sA[(size_t)(kc * 32 + ln) * 16 + idx] = f2bf(v);
    }
    __syncthreads();
    int nkc = ks >> 5;
    for (int kc = 0; kc < nkc; ++kc) {
      v16bf av = *reinterpret_cast<const v16bf*>(&sA[(size_t)(kc * 32 + lane) * 16]);
      int kcg = (k0 >> 5) + kc;
      const unsigned short* wbase = Wp + (size_t)kcg * NCH * 512 + lane * 16;
#pragma unroll
      for (int ci = 0; ci < NCPW; ++ci) {
        int nc = wave + ci * 4;
        if (nc < NCH) {
          v16bf bv = *reinterpret_cast<const v16bf*>(wbase + (size_t)nc * 512);
          acc[ci] = __builtin_amdgcn_wmma_f32_16x16x32_bf16(
              false, av, false, bv, (short)0, acc[ci], false, false);
        }
      }
    }
    __syncthreads();
  }
  {
#pragma unroll
    for (int ci = 0; ci < NCPW; ++ci) {
      int nc = wave + ci * 4;
      if (nc < NCH) {
        int n = (lane & 15) + (nc << 4);
        int mhi = (lane >> 4) << 3;
#pragma unroll
        for (int r = 0; r < 8; ++r) sC[(r + mhi) * NOUT + n] = acc[ci][r];
      }
    }
  }
  __syncthreads();
  for (int t = tid; t < 16 * NOUT; t += 128) {
    int m = t / NOUT, n = t % NOUT;
    int rg = row0 + m;
    float v = sC[t] + E_.bias[n];
    if (E_.relu) v = v > 0.f ? v : 0.f;
    if (E_.rscale) v *= E_.rscale[rg];
    if (E_.resid) v += E_.resid[(size_t)rg * NOUT + n];
    sC[t] = v;
  }
  __syncthreads();
  if (E_.ln_g) {
    {
      int m = tid >> 3, sl = tid & 7;
      constexpr int per = NOUT >> 3;
      float s = 0.f, s2 = 0.f;
      for (int j = 0; j < per; ++j) {
        float v = sC[m * NOUT + sl * per + j];
        s += v; s2 += v * v;
      }
      sSum[m * 8 + sl] = s; sSq[m * 8 + sl] = s2;
    }
    __syncthreads();
    if (tid < 16) {
      float s = 0.f, s2 = 0.f;
      for (int j = 0; j < 8; ++j) { s += sSum[tid * 8 + j]; s2 += sSq[tid * 8 + j]; }
      float mean = s / NOUT;
      float var = s2 / NOUT - mean * mean;
      sMean[tid] = mean; sInv[tid] = rsqrtf(var + 1e-5f);
    }
    __syncthreads();
  }
  for (int t = tid; t < 16 * NOUT; t += 128) {
    int m = t / NOUT, n = t % NOUT;
    size_t og = (size_t)(row0 + m) * NOUT + n;
    float v = sC[t];
    if (E_.ln_g) v = (v - sMean[m]) * sInv[m] * E_.ln_g[n] + E_.ln_b[n];
    if (E_.out_f) E_.out_f[og] = v;
    if (E_.out_h) E_.out_h[og] = f2bf(v);
  }
}

// Pack W [K,N] fp32 row-major -> WMMA-B packed bf16 (Kpad padded with zeros).
__global__ void k_pack_w(const float* W, unsigned short* dst, int K, int Kpad, int N) {
  int total = Kpad * N;
  for (int t = blockIdx.x * blockDim.x + threadIdx.x; t < total; t += gridDim.x * blockDim.x) {
    int idx = t & 15, lane = (t >> 4) & 31, blk = t >> 9;
    int nchunks = N >> 4;
    int kc = blk / nchunks, nc = blk % nchunks;
    int g = lane >> 4, nl = lane & 15;
    int kl = ((idx >> 3) << 4) + g * 8 + (idx & 7);
    int k = kc * 32 + kl, n = nc * 16 + nl;
    float v = (k < K) ? W[(size_t)k * N + n] : 0.f;
    dst[t] = f2bf(v);
  }
}

// ---------------- elementwise / segment kernels ----------------
__global__ void k_fill_f32(float* p, float v, int n) {
  int t = blockIdx.x * blockDim.x + threadIdx.x; if (t < n) p[t] = v;
}
__global__ void k_fill_u32(unsigned int* p, unsigned int v, int n) {
  int t = blockIdx.x * blockDim.x + threadIdx.x; if (t < n) p[t] = v;
}
__global__ void k_small_lin(const float* X, const float* W, const float* B, float* out,
                            int M, int K, int Nout, const float* rsin, const float* rsout) {
  int t = blockIdx.x * blockDim.x + threadIdx.x;
  if (t >= M * Nout) return;
  int m = t / Nout, n = t % Nout;
  const float* x = X + (size_t)m * K;
  float acc = 0.f;
  for (int i = 0; i < K; ++i) acc += x[i] * W[(size_t)i * Nout + n];
  if (rsin) acc *= rsin[m];
  acc += B[n];
  if (rsout) acc *= rsout[m];
  out[t] = acc;
}
__global__ void k_xform_pts(float* pts, const float* rot, const float* trans, int npp, int total) {
  int t = blockIdx.x * blockDim.x + threadIdx.x;
  if (t >= total) return;
  int n = t / npp, i = t % npp;
  float* p = pts + ((size_t)n * npp + i) * 3;
  const float* R = rot + (size_t)n * 9;
  const float* T = trans + (size_t)n * 3;
  float x = p[0], y = p[1], z = p[2];
  p[0] = R[0]*x + R[1]*y + R[2]*z + T[0];
  p[1] = R[3]*x + R[4]*y + R[5]*z + T[1];
  p[2] = R[6]*x + R[7]*y + R[8]*z + T[2];
}
__global__ void k_logits(const float* q, const float* k, const float* qp, const float* kp,
                         const float* b, const float* head_w, const float* mask,
                         const int* er, const int* ec, float* lg, unsigned int* mx, int ne) {
  int t = blockIdx.x * blockDim.x + threadIdx.x;
  if (t >= ne * 8) return;
  int e = t >> 3, h = t & 7;
  int r = er[e], c = ec[e];
  const float* qr = q + (size_t)r * 128 + h * 16;
  const float* kc_ = k + (size_t)c * 128 + h * 16;
  float qk = 0.f;
#pragma unroll
  for (int i = 0; i < 16; ++i) qk += qr[i] * kc_[i];
  qk *= 0.25f;  // 1/sqrt(C_HIDDEN)
  const float* qpr = qp + (size_t)r * 192 + h * 24;
  const float* kpc = kp + (size_t)c * 192 + h * 24;
  float d2 = 0.f;
#pragma unroll
  for (int j = 0; j < 24; ++j) { float d = qpr[j] - kpc[j]; d2 += d * d; }
  float g = log1pf(expf(head_w[h]));           // softplus
  const float wl = 0.5773502691896258f;        // sqrt(1/3)
  const float wc = 1.0f / 6.0f;                // sqrt(2/(9*8))
  float lgt = wl * (qk + b[t]) - 0.5f * wl * wc * g * d2 + (mask[c] - 1.f) * 1e5f;
  lg[t] = lgt;
  atomicMax(&mx[r * 8 + h], encf(lgt));
}
__global__ void k_fixmx(unsigned int* mx, int n) {
  int t = blockIdx.x * blockDim.x + threadIdx.x; if (t >= n) return;
  float v = decf(mx[t]);
  if (!isfinite(v)) v = 0.f;
  ((float*)mx)[t] = v;
}
__global__ void k_exp(float* lg, const float* mxf, float* den, const int* er, int ne) {
  int t = blockIdx.x * blockDim.x + threadIdx.x; if (t >= ne * 8) return;
  int e = t >> 3, h = t & 7;
  float ex = expf(lg[t] - mxf[er[e] * 8 + h]);
  lg[t] = ex;
  atomicAdd(&den[er[e] * 8 + h], ex);
}
__global__ void k_norm(float* lg, const float* den, const int* er, int ne) {
  int t = blockIdx.x * blockDim.x + threadIdx.x; if (t >= ne * 8) return;
  int e = t >> 3, h = t & 7;
  lg[t] = lg[t] / (den[er[e] * 8 + h] + 1e-9f);
}
__global__ void k_scatter(const float* a, const float* v, const float* vp, const float* z,
                          const int* er, const int* ec, float* o, float* optg, float* opair, int ne) {
  int t = blockIdx.x * blockDim.x + threadIdx.x; if (t >= ne * 8) return;
  int e = t >> 3, h = t & 7;
  int r = er[e], c = ec[e];
  float w = a[t];
  const float* vc = v + (size_t)c * 128 + h * 16;
  float* orr = o + (size_t)r * 128 + h * 16;
#pragma unroll
  for (int i = 0; i < 16; ++i) atomicAdd(&orr[i], w * vc[i]);
  const float* vpc = vp + (size_t)c * 288 + h * 36;
  float* og = optg + (size_t)r * 288 + h * 36;
#pragma unroll
  for (int j = 0; j < 36; ++j) atomicAdd(&og[j], w * vpc[j]);
  const float* ze = z + (size_t)e * 128;
  float* op = opair + (size_t)r * 1024 + h * 128;
  for (int j = 0; j < 128; ++j) atomicAdd(&op[j], w * ze[j]);
}
// Build per-node IPA feature vector [1536] (bf16), rotating points back to local frame.
__global__ __launch_bounds__(128) void k_feats(const float* o, const float* optg, const float* opair,
                                               const float* rot, const float* trans,
                                               unsigned short* feats) {
  int n = blockIdx.x, tid = threadIdx.x;
  const float* R = rot + (size_t)n * 9;
  const float* T = trans + (size_t)n * 3;
  unsigned short* F = feats + (size_t)n * 1536;
  for (int j = tid; j < 128; j += 128) F[j] = f2bf(o[(size_t)n * 128 + j]);
  for (int hp = tid; hp < 96; hp += 128) {
    const float* g = optg + (size_t)n * 288 + hp * 3;
    float gx = g[0] - T[0], gy = g[1] - T[1], gz = g[2] - T[2];
    float lx = R[0]*gx + R[3]*gy + R[6]*gz;   // R^T
    float ly = R[1]*gx + R[4]*gy + R[7]*gz;
    float lz = R[2]*gx + R[5]*gy + R[8]*gz;
    F[128 + hp * 3 + 0] = f2bf(lx);
    F[128 + hp * 3 + 1] = f2bf(ly);
    F[128 + hp * 3 + 2] = f2bf(lz);
    F[416 + hp] = f2bf(sqrtf(lx*lx + ly*ly + lz*lz + 1e-8f));
  }
  for (int j = tid; j < 1024; j += 128) F[512 + j] = f2bf(opair[(size_t)n * 1024 + j]);
}
__global__ void k_mask_node(float* node, float* outn, const float* mask, int n256) {
  int t = blockIdx.x * blockDim.x + threadIdx.x; if (t >= n256) return;
  float v = node[t] * mask[t >> 8];
  node[t] = v; outn[t] = v;
}
__global__ void k_compose(const float* rot, const float* trans, const float* u6,
                          float* ro, float* to, int n) {
  int t = blockIdx.x * blockDim.x + threadIdx.x; if (t >= n) return;
  const float* u = u6 + (size_t)t * 6;
  float qw = 1.f, qx = u[0], qy = u[1], qz = u[2];
  float inv = rsqrtf(qw*qw + qx*qx + qy*qy + qz*qz);
  qw *= inv; qx *= inv; qy *= inv; qz *= inv;
  float Q[9] = {1-2*(qy*qy+qz*qz), 2*(qx*qy-qw*qz), 2*(qx*qz+qw*qy),
                2*(qx*qy+qw*qz), 1-2*(qx*qx+qz*qz), 2*(qy*qz-qw*qx),
                2*(qx*qz-qw*qy), 2*(qy*qz+qw*qx), 1-2*(qx*qx+qy*qy)};
  const float* R = rot + (size_t)t * 9;
  float* Ro = ro + (size_t)t * 9;
#pragma unroll
  for (int i = 0; i < 3; ++i)
#pragma unroll
    for (int k = 0; k < 3; ++k)
      Ro[i*3+k] = R[i*3+0]*Q[0*3+k] + R[i*3+1]*Q[1*3+k] + R[i*3+2]*Q[2*3+k];
  const float* T = trans + (size_t)t * 3;
  float* To = to + (size_t)t * 3;
#pragma unroll
  for (int i = 0; i < 3; ++i)
    To[i] = T[i] + R[i*3+0]*u[3] + R[i*3+1]*u[4] + R[i*3+2]*u[5];
}
// y (bf16 [ne,384]) += concat(seqz[e], h[s0[e]], h[s1[e]])
__global__ void k_addx(unsigned short* y, const float* seqz, const float* h,
                       const int* s0, const int* s1, int ne) {
  int t = blockIdx.x * blockDim.x + threadIdx.x; if (t >= ne * 384) return;
  int e = t / 384, j = t % 384;
  float x;
  if (j < 128)      x = seqz[(size_t)e * 128 + j];
  else if (j < 256) x = h[(size_t)s0[e] * 128 + (j - 128)];
  else              x = h[(size_t)s1[e] * 128 + (j - 256)];
  y[t] = f2bf(bf2f(y[t]) + x);
}

// ---------------- host orchestration ----------------
struct IPAOff {
  size_t bias_b, bias_w, head_w, k_b, k_w, kp_b, kp_w, out_b, out_w,
         q_b, q_w, qp_b, qp_w, v_b, v_w, vp_b, vp_w;
};

extern "C" void kernel_launch(void* const* d_in, const int* in_sizes, int n_in,
                              void* d_out, int out_size, void* d_ws, size_t ws_size,
                              hipStream_t stream) {
  (void)in_sizes; (void)n_in; (void)out_size; (void)ws_size;
  auto cdiv = [](long long a, long long b) { return (int)((a + b - 1) / b); };

  const float* node_in  = (const float*)d_in[0];
  const float* rot_in   = (const float*)d_in[1];
  const float* trans_in = (const float*)d_in[2];
  const float* lat_in   = (const float*)d_in[3];
  const float* edge_in  = (const float*)d_in[4];   // [E,132]
  const float* seq_in   = (const float*)d_in[5];   // [ES,132]
  const float* seqf_in  = (const float*)d_in[6];   // [ES,128]
  const int*   eidx     = (const int*)d_in[7];     // [2,E]
  const int*   sidx     = (const int*)d_in[8];     // [2,ES]
  const float* mask     = (const float*)d_in[9];
  const float* noising  = (const float*)d_in[10];
  const float* P        = (const float*)d_in[11];  // flat params (pytree alpha order)

  const int *er = eidx, *ec = eidx + EE;
  const int *sr = sidx, *sc = sidx + ES;

  float* out = (float*)d_out;
  float* out_node  = out;
  float* out_rot   = out_node + (size_t)NN * 256;
  float* out_trans = out_rot + (size_t)NN * 9;
  float* out_edge  = out_trans + (size_t)NN * 3;
  float* out_seq   = out_edge + (size_t)EE * 128;
  float* out_lat   = out_seq + (size_t)ES * 128;

  // ---- parameter offsets (jax pytree flatten: dict keys sorted alphabetically) ----
  size_t off = 0;
  auto take = [&](size_t n) { size_t o = off; off += n; return o; };
  size_t bb_b = take(6), bb_w = take(256 * 6);
  size_t ee_l1_b = take(128), ee_l1_w = take(644 * 128);
  size_t ee_l2_b = take(128), ee_l2_w = take(128 * 128);
  size_t ee_l3_b = take(128), ee_l3_w = take(128 * 128);
  size_t ee_ln_b = take(128), ee_ln_g = take(128);
  size_t et_fin_b = take(128), et_fin_w = take(384 * 128);
  size_t et_init_b = take(128), et_init_w = take(256 * 128);
  size_t et_ln_b = take(128), et_ln_g = take(128);
  size_t et_t1_b = take(384), et_t1_w = take(384 * 384);
  size_t et_t2_b = take(384), et_t2_w = take(384 * 384);
  auto ipa_off = [&]() {
    IPAOff o;
    o.bias_b = take(8); o.bias_w = take(128 * 8);
    o.head_w = take(8);
    o.k_b = take(128); o.k_w = take(256 * 128);
    o.kp_b = take(192); o.kp_w = take(256 * 192);
    o.out_b = take(256); o.out_w = take(1536 * 256);
    o.q_b = take(128); o.q_w = take(256 * 128);
    o.qp_b = take(192); o.qp_w = take(256 * 192);
    o.v_b = take(128); o.v_w = take(256 * 128);
    o.vp_b = take(288); o.vp_w = take(256 * 288);
    return o;
  };
  IPAOff iseq = ipa_off();  // 'ipa_seq' < 'ipa_sp'
  IPAOff isp  = ipa_off();
  size_t l2n_l1_b = take(256), l2n_l1_w = take(512 * 256);
  size_t l2n_l2_b = take(256), l2n_l2_w = take(256 * 256);
  size_t l2n_l3_b = take(256), l2n_l3_w = take(256 * 256);
  size_t lns0_b = take(256), lns0_g = take(256);
  size_t lns1_b = take(256), lns1_g = take(256);
  size_t lns2_b = take(256), lns2_g = take(256);
  size_t n2l_l1_b = take(256), n2l_l1_w = take(512 * 256);
  size_t n2l_l2_b = take(256), n2l_l2_w = take(256 * 256);
  size_t n2l_l3_b = take(256), n2l_l3_w = take(256 * 256);
  size_t nt_l1_b = take(256), nt_l1_w = take(256 * 256);
  size_t nt_l2_b = take(256), nt_l2_w = take(256 * 256);
  size_t nt_l3_b = take(256), nt_l3_w = take(256 * 256);
  size_t nt_ln_b = take(256), nt_ln_g = take(256);
  size_t sel_b = take(128), sel_g = take(128);
  size_t su_l1_b = take(128), su_l1_w = take(644 * 128);
  size_t su_l2_b = take(128), su_l2_w = take(128 * 128);
  size_t su_l3_b = take(128), su_l3_w = take(128 * 128);

  // ---- workspace allocator ----
  char* wsp = (char*)d_ws;
  size_t wused = 0;
  auto alloc = [&](size_t bytes) -> void* {
    size_t a = (bytes + 255) & ~(size_t)255;
    void* p = wsp + wused; wused += a; return p;
  };

  // ---- pack all GEMM weights to WMMA-B bf16 layout ----
  auto pack = [&](size_t woff, int K, int Kpad, int N) -> const unsigned short* {
    unsigned short* dst = (unsigned short*)alloc((size_t)Kpad * N * 2);
    int total = Kpad * N;
    k_pack_w<<<cdiv(total, 256), 256, 0, stream>>>(P + woff, dst, K, Kpad, N);
    return dst;
  };
  const unsigned short* Wee1 = pack(ee_l1_w, 644, 672, 128);
  const unsigned short* Wee2 = pack(ee_l2_w, 128, 128, 128);
  const unsigned short* Wee3 = pack(ee_l3_w, 128, 128, 128);
  const unsigned short* Wsu1 = pack(su_l1_w, 644, 672, 128);
  const unsigned short* Wsu2 = pack(su_l2_w, 128, 128, 128);
  const unsigned short* Wsu3 = pack(su_l3_w, 128, 128, 128);
  const unsigned short* Wl2n1 = pack(l2n_l1_w, 512, 512, 256);
  const unsigned short* Wl2n2 = pack(l2n_l2_w, 256, 256, 256);
  const unsigned short* Wl2n3 = pack(l2n_l3_w, 256, 256, 256);
  const unsigned short *Wq[2], *Wk[2], *Wv[2], *Wqp[2], *Wkp[2], *Wvp[2], *Wout[2];
  const IPAOff* IP[2] = { &isp, &iseq };
  for (int s = 0; s < 2; ++s) {
    Wq[s]  = pack(IP[s]->q_w,  256, 256, 128);
    Wk[s]  = pack(IP[s]->k_w,  256, 256, 128);
    Wv[s]  = pack(IP[s]->v_w,  256, 256, 128);
    Wqp[s] = pack(IP[s]->qp_w, 256, 256, 192);
    Wkp[s] = pack(IP[s]->kp_w, 256, 256, 192);
    Wvp[s] = pack(IP[s]->vp_w, 256, 256, 288);
    Wout[s] = pack(IP[s]->out_w, 1536, 1536, 256);
  }
  const unsigned short* Wnt1 = pack(nt_l1_w, 256, 256, 256);
  const unsigned short* Wnt2 = pack(nt_l2_w, 256, 256, 256);
  const unsigned short* Wnt3 = pack(nt_l3_w, 256, 256, 256);
  const unsigned short* Weti = pack(et_init_w, 256, 256, 128);
  const unsigned short* Wet1 = pack(et_t1_w, 384, 384, 384);
  const unsigned short* Wet2 = pack(et_t2_w, 384, 384, 384);
  const unsigned short* Wetf = pack(et_fin_w, 384, 384, 128);
  const unsigned short* Wn2l1 = pack(n2l_l1_w, 512, 512, 256);
  const unsigned short* Wn2l2 = pack(n2l_l2_w, 256, 256, 256);
  const unsigned short* Wn2l3 = pack(n2l_l3_w, 256, 256, 256);

  // ---- activation buffers ----
  size_t hmax = (size_t)ES * 384;                       // also covers E*128
  if ((size_t)EE * 128 > hmax) hmax = (size_t)EE * 128;
  unsigned short* h1 = (unsigned short*)alloc(hmax * 2);
  unsigned short* h2 = (unsigned short*)alloc(hmax * 2);
  float* seq_z    = (float*)alloc((size_t)ES * 128 * 4);
  float* node_cur = (float*)alloc((size_t)NN * 256 * 4);
  float* qb = (float*)alloc((size_t)NN * 128 * 4);
  float* kb = (float*)alloc((size_t)NN * 128 * 4);
  float* vb = (float*)alloc((size_t)NN * 128 * 4);
  float* qp = (float*)alloc((size_t)NN * 192 * 4);
  float* kp = (float*)alloc((size_t)NN * 192 * 4);
  float* vp = (float*)alloc((size_t)NN * 288 * 4);
  float* bbuf = (float*)alloc((size_t)EE * 8 * 4);
  float* lg   = (float*)alloc((size_t)EE * 8 * 4);
  unsigned int* mx = (unsigned int*)alloc((size_t)NN * 8 * 4);
  float* den  = (float*)alloc((size_t)NN * 8 * 4);
  float* obuf = (float*)alloc((size_t)NN * 128 * 4);
  float* optg = (float*)alloc((size_t)NN * 288 * 4);
  float* opair = (float*)alloc((size_t)NN * 1024 * 4);
  unsigned short* feats = (unsigned short*)alloc((size_t)NN * 1536 * 2);
  float* hinit = (float*)alloc((size_t)NN * 128 * 4);
  float* upd6  = (float*)alloc((size_t)NN * 6 * 4);

  // ---- GEMM helpers ----
  auto A1 = [](const void* s, int w, int ty, const int* ix = nullptr) {
    ASpec a{}; a.s0 = s; a.w0 = w; a.t0 = ty; a.i0 = ix; return a;
  };
  auto A3 = [](const void* s0, int w0, int t0, const int* i0,
               const void* s1, int w1, int t1, const int* i1,
               const void* s2, int w2, int t2, const int* i2) {
    ASpec a{}; a.s0 = s0; a.w0 = w0; a.t0 = t0; a.i0 = i0;
    a.s1 = s1; a.w1 = w1; a.t1 = t1; a.i1 = i1;
    a.s2 = s2; a.w2 = w2; a.t2 = t2; a.i2 = i2; return a;
  };
  auto mkE = [](const float* bias, int relu, const float* resid, const float* rscale,
                const float* lng, const float* lnb, float* of, unsigned short* oh) {
    ESpec e{}; e.bias = bias; e.relu = relu; e.resid = resid; e.rscale = rscale;
    e.ln_g = lng; e.ln_b = lnb; e.out_f = of; e.out_h = oh; return e;
  };
  auto gemm = [&](ASpec a, ESpec e, const unsigned short* Wp, int M, int Kpad, int Nout) {
    size_t smem = (size_t)GKS * 16 * 2 + (size_t)16 * Nout * 4 + 16 * 8 * 4 * 2 + 16 * 4 * 2;
    dim3 grid(M / 16), blk(128);
    switch (Nout) {
      case 128: k_gemm<128><<<grid, blk, smem, stream>>>(a, e, Wp, M, Kpad); break;
      case 192: k_gemm<192><<<grid, blk, smem, stream>>>(a, e, Wp, M, Kpad); break;
      case 256: k_gemm<256><<<grid, blk, smem, stream>>>(a, e, Wp, M, Kpad); break;
      case 288: k_gemm<288><<<grid, blk, smem, stream>>>(a, e, Wp, M, Kpad); break;
      case 384: k_gemm<384><<<grid, blk, smem, stream>>>(a, e, Wp, M, Kpad); break;
    }
  };

  // ---- 1. edge_embed -> out_edge ----
  gemm(A3(edge_in, 132, 0, nullptr, node_in, 256, 0, er, node_in, 256, 0, ec),
       mkE(P + ee_l1_b, 1, 0, 0, 0, 0, nullptr, h1), Wee1, EE, 672, 128);
  gemm(A1(h1, 128, 1), mkE(P + ee_l2_b, 1, 0, 0, 0, 0, nullptr, h2), Wee2, EE, 128, 128);
  gemm(A1(h2, 128, 1), mkE(P + ee_l3_b, 0, 0, 0, P + ee_ln_g, P + ee_ln_b, out_edge, nullptr),
       Wee3, EE, 128, 128);

  // ---- 2. seq_edge_update -> seq_z ----
  gemm(A3(seq_in, 132, 0, nullptr, node_in, 256, 0, sr, node_in, 256, 0, sc),
       mkE(P + su_l1_b, 1, 0, 0, 0, 0, nullptr, h1), Wsu1, ES, 672, 128);
  gemm(A1(h1, 128, 1), mkE(P + su_l2_b, 1, 0, 0, 0, 0, nullptr, h2), Wsu2, ES, 128, 128);
  gemm(A1(h2, 128, 1), mkE(P + su_l3_b, 0, seqf_in, 0, P + sel_g, P + sel_b, seq_z, nullptr),
       Wsu3, ES, 128, 128);

  // ---- 3. lat2node -> node_cur ----
  gemm(A3(lat_in, 256, 0, nullptr, node_in, 256, 0, nullptr, nullptr, 0, 0, nullptr),
       mkE(P + l2n_l1_b, 1, 0, 0, 0, 0, nullptr, h1), Wl2n1, NN, 512, 256);
  gemm(A1(h1, 256, 1), mkE(P + l2n_l2_b, 1, 0, 0, 0, 0, nullptr, h2), Wl2n2, NN, 256, 256);
  gemm(A1(h2, 256, 1), mkE(P + l2n_l3_b, 0, node_in, 0, P + lns0_g, P + lns0_b, node_cur, nullptr),
       Wl2n3, NN, 256, 256);

  // ---- 4/5. IPA stages ----
  auto run_ipa = [&](int s, const IPAOff& ip, const float* z, const int* r_, const int* c_,
                     int ne, const float* lng, const float* lnb) {
    gemm(A1(node_cur, 256, 0), mkE(P + ip.q_b, 0, 0, 0, 0, 0, qb, nullptr), Wq[s], NN, 256, 128);
    gemm(A1(node_cur, 256, 0), mkE(P + ip.k_b, 0, 0, 0, 0, 0, kb, nullptr), Wk[s], NN, 256, 128);
    gemm(A1(node_cur, 256, 0), mkE(P + ip.v_b, 0, 0, 0, 0, 0, vb, nullptr), Wv[s], NN, 256, 128);
    gemm(A1(node_cur, 256, 0), mkE(P + ip.qp_b, 0, 0, 0, 0, 0, qp, nullptr), Wqp[s], NN, 256, 192);
    gemm(A1(node_cur, 256, 0), mkE(P + ip.kp_b, 0, 0, 0, 0, 0, kp, nullptr), Wkp[s], NN, 256, 192);
    gemm(A1(node_cur, 256, 0), mkE(P + ip.vp_b, 0, 0, 0, 0, 0, vp, nullptr), Wvp[s], NN, 256, 288);
    k_xform_pts<<<cdiv((long long)NN * 64, 256), 256, 0, stream>>>(qp, rot_in, trans_in, 64, NN * 64);
    k_xform_pts<<<cdiv((long long)NN * 64, 256), 256, 0, stream>>>(kp, rot_in, trans_in, 64, NN * 64);
    k_xform_pts<<<cdiv((long long)NN * 96, 256), 256, 0, stream>>>(vp, rot_in, trans_in, 96, NN * 96);
    k_small_lin<<<cdiv((long long)ne * 8, 256), 256, 0, stream>>>(
        z, P + ip.bias_w, P + ip.bias_b, bbuf, ne, 128, 8, nullptr, nullptr);
    k_fill_u32<<<cdiv(NN * 8, 256), 256, 0, stream>>>(mx, 0u, NN * 8);
    k_fill_f32<<<cdiv(NN * 8, 256), 256, 0, stream>>>(den, 0.f, NN * 8);
    k_fill_f32<<<cdiv(NN * 128, 256), 256, 0, stream>>>(obuf, 0.f, NN * 128);
    k_fill_f32<<<cdiv(NN * 288, 256), 256, 0, stream>>>(optg, 0.f, NN * 288);
    k_fill_f32<<<cdiv(NN * 1024, 256), 256, 0, stream>>>(opair, 0.f, NN * 1024);
    k_logits<<<cdiv((long long)ne * 8, 256), 256, 0, stream>>>(
        qb, kb, qp, kp, bbuf, P + ip.head_w, mask, r_, c_, lg, mx, ne);
    k_fixmx<<<cdiv(NN * 8, 256), 256, 0, stream>>>(mx, NN * 8);
    k_exp<<<cdiv((long long)ne * 8, 256), 256, 0, stream>>>(lg, (const float*)mx, den, r_, ne);
    k_norm<<<cdiv((long long)ne * 8, 256), 256, 0, stream>>>(lg, den, r_, ne);
    k_scatter<<<cdiv((long long)ne * 8, 256), 256, 0, stream>>>(
        lg, vb, vp, z, r_, c_, obuf, optg, opair, ne);
    k_feats<<<dim3(NN), dim3(128), 0, stream>>>(obuf, optg, opair, rot_in, trans_in, feats);
    gemm(A1(feats, 1536, 1), mkE(P + ip.out_b, 0, node_cur, mask, lng, lnb, node_cur, nullptr),
         Wout[s], NN, 1536, 256);
  };
  run_ipa(0, isp, out_edge, er, ec, EE, P + lns1_g, P + lns1_b);
  run_ipa(1, iseq, seq_z, sr, sc, ES, P + lns2_g, P + lns2_b);

  // ---- 6. node_trans + mask -> node_cur, out_node ----
  gemm(A1(node_cur, 256, 0), mkE(P + nt_l1_b, 1, 0, 0, 0, 0, nullptr, h1), Wnt1, NN, 256, 256);
  gemm(A1(h1, 256, 1), mkE(P + nt_l2_b, 1, 0, 0, 0, 0, nullptr, h2), Wnt2, NN, 256, 256);
  gemm(A1(h2, 256, 1), mkE(P + nt_l3_b, 0, node_cur, 0, P + nt_ln_g, P + nt_ln_b, node_cur, nullptr),
       Wnt3, NN, 256, 256);
  k_mask_node<<<cdiv(NN * 256, 256), 256, 0, stream>>>(node_cur, out_node, mask, NN * 256);

  // ---- 7. backbone update -> out_rot, out_trans ----
  k_small_lin<<<cdiv(NN * 6, 256), 256, 0, stream>>>(
      node_cur, P + bb_w, P + bb_b, upd6, NN, 256, 6, noising, noising);
  k_compose<<<cdiv(NN, 256), 256, 0, stream>>>(rot_in, trans_in, upd6, out_rot, out_trans, NN);

  // ---- 8. edge_transition -> out_seq ----
  gemm(A1(node_cur, 256, 0), mkE(P + et_init_b, 0, 0, 0, 0, 0, hinit, nullptr), Weti, NN, 256, 128);
  gemm(A3(seq_z, 128, 0, nullptr, hinit, 128, 0, sr, hinit, 128, 0, sc),
       mkE(P + et_t1_b, 1, 0, 0, 0, 0, nullptr, h1), Wet1, ES, 384, 384);
  gemm(A1(h1, 384, 1), mkE(P + et_t2_b, 1, 0, 0, 0, 0, nullptr, h2), Wet2, ES, 384, 384);
  k_addx<<<cdiv((long long)ES * 384, 256), 256, 0, stream>>>(h2, seq_z, hinit, sr, sc, ES);
  gemm(A1(h2, 384, 1), mkE(P + et_fin_b, 0, 0, 0, P + et_ln_g, P + et_ln_b, out_seq, nullptr),
       Wetf, ES, 384, 128);

  // ---- 9. node2lat -> out_lat ----
  gemm(A3(lat_in, 256, 0, nullptr, node_cur, 256, 0, nullptr, nullptr, 0, 0, nullptr),
       mkE(P + n2l_l1_b, 1, 0, 0, 0, 0, nullptr, h1), Wn2l1, NN, 512, 256);
  gemm(A1(h1, 256, 1), mkE(P + n2l_l2_b, 1, 0, 0, 0, 0, nullptr, h2), Wn2l2, NN, 256, 256);
  gemm(A1(h2, 256, 1), mkE(P + n2l_l3_b, 0, lat_in, 0, 0, 0, out_lat, nullptr), Wn2l3, NN, 256, 256);
}
